// MCKT_6571299963512
// MI455X (gfx1250) — matrix operand order repaired and checked by reading, not compiled, not run on hardware
//
#include <hip/hip_runtime.h>
#include <math.h>

#define B_  8
#define S_  1024
#define D_  1024
#define H_  8
#define DK_ 128
#define NEGV -1000000000.0f

typedef __bf16 bf16_t;
typedef bf16_t v16bf __attribute__((ext_vector_type(16)));
typedef float  v8f   __attribute__((ext_vector_type(8)));

#define BM 128
#define BN 128
#define BK 32
#define LST 56          // LDS row stride (elems): 112B rows -> 16B aligned, conflict-free frag reads
#define NTHREADS 128    // 4 waves (wave32), each computes a 64x64 sub-tile

#if __has_builtin(__builtin_amdgcn_global_load_async_to_lds_b128) && __has_builtin(__builtin_amdgcn_s_wait_asynccnt)
#define HAVE_ASYNC 1
#else
#define HAVE_ASYNC 0
#endif

// Types matching the ROCm builtin signature: v4i in AS1 (global) / AS3 (LDS)
typedef int vint4 __attribute__((vector_size(4 * sizeof(int))));
typedef __attribute__((address_space(1))) vint4 g_vint4;
typedef __attribute__((address_space(3))) vint4 l_vint4;

__device__ __forceinline__ unsigned short f32_to_bf16_rne(float f) {
  union { bf16_t b; unsigned short u; } c; c.b = (bf16_t)f; return c.u;  // native v_cvt on gfx1250
}
__device__ __forceinline__ unsigned int pack_bf16x2(float lo, float hi) {
  return (unsigned int)f32_to_bf16_rne(lo) | ((unsigned int)f32_to_bf16_rne(hi) << 16);
}

// 16B global -> LDS copy: async (ASYNCcnt) when available, sync uint4 otherwise.
__device__ __forceinline__ void cp16_g2l(const unsigned short* g, unsigned short* l) {
#if HAVE_ASYNC
  __builtin_amdgcn_global_load_async_to_lds_b128((g_vint4*)g, (l_vint4*)l, 0, 0);
#else
  *(uint4*)l = *(const uint4*)g;
#endif
}
__device__ __forceinline__ void wait_async_prev() {  // <=8 outstanding: next tile may be in flight
#if HAVE_ASYNC
  __builtin_amdgcn_s_wait_asynccnt(8);
#endif
}
__device__ __forceinline__ void wait_async_all() {
#if HAVE_ASYNC
  __builtin_amdgcn_s_wait_asynccnt(0);
#endif
}

union FragU { v16bf v; unsigned short u[16]; };

// A fragment 16x32 bf16 (ISA 7.12.2): lane holds row M=lane&15; elems[0..7]=K kb..kb+7,
// elems[8..15]=K kb+16..kb+23, kb=(lane>>4)*8.  Two 16B LDS reads.
__device__ __forceinline__ v16bf load_frag_a(const unsigned short* aLds, int row0, int lane) {
  const unsigned short* p = aLds + (size_t)(row0 + (lane & 15)) * LST + ((lane >> 4) << 3);
  FragU f;
#pragma unroll
  for (int i = 0; i < 8; ++i) { f.u[i] = p[i]; f.u[8 + i] = p[16 + i]; }
  return f.v;
}
// B fragment 32x16 bf16: lane holds col n=lane&15; elems[0..7]=K kb..kb+7,
// elems[8..15]=K kb+16..kb+23, kb=(lane>>4)*8.  bLds is [n][k] (two 16B reads).
__device__ __forceinline__ v16bf load_frag_b(const unsigned short* bLds, int col0, int lane) {
  const unsigned short* p = bLds + (size_t)(col0 + (lane & 15)) * LST + ((lane >> 4) << 3);
  FragU f;
#pragma unroll
  for (int i = 0; i < 8; ++i) { f.u[i] = p[i]; f.u[8 + i] = p[16 + i]; }
  return f.v;
}

// MODE 0: proj   A f32 [m][k], B f32 W[n][k], +bias, out bf16 -> [B,H,S,DK]
// MODE 1: scores A bf16 Q_bh, B bf16 K_bh[n][k]; out f32 -> attn; causal tile skip; async double-buffer
// MODE 2: attn@V A f32 attn_bh, B bf16 V_bh[k][n] (transpose staging), out bf16 ctx [B,S,D]; causal kEnd
// MODE 3: oproj  A bf16 ctx (async), B f32 Wo[n][k], +bias, out f32
template<int MODE>
__global__ void __launch_bounds__(NTHREADS) gemm_wmma_kernel(
    const void* __restrict__ Asrc, int lda, const void* __restrict__ Bsrc, int ldb,
    const float* __restrict__ bias, void* __restrict__ Cdst, int K, float alpha) {

  constexpr int NBUF = (MODE == 1) ? 2 : 1;
  __shared__ unsigned short aLds[NBUF * BM * LST];
  __shared__ unsigned short bLds[NBUF * BN * LST];

  const int tid  = threadIdx.x;
  const int lane = tid & 31;
  const int wave = tid >> 5;
  const int m0 = blockIdx.y * BM;
  const int n0 = blockIdx.x * BN;
  const int bh = blockIdx.z;

  if constexpr (MODE == 1) {            // tile strictly above diagonal -> fully masked
    if (n0 > m0 + BM - 1) return;
  }
  int kEnd = K;
  if constexpr (MODE == 2) kEnd = (K < m0 + BM) ? K : (m0 + BM);  // attn[s,t]==0 for t>s

  const float*          Af = (const float*)Asrc;
  const unsigned short* Ah = (const unsigned short*)Asrc;
  const float*          Bf = (const float*)Bsrc;
  const unsigned short* Bh = (const unsigned short*)Bsrc;
  if constexpr (MODE == 1) { Ah += (size_t)bh * S_ * DK_; Bh += (size_t)bh * S_ * DK_; }
  if constexpr (MODE == 2) { Af += (size_t)bh * S_ * S_;  Bh += (size_t)bh * S_ * DK_; }

  const int wm = (wave >> 1) * 64;
  const int wn = (wave & 1) * 64;

  v8f acc[4][4];
#pragma unroll
  for (int i = 0; i < 4; ++i)
#pragma unroll
    for (int j = 0; j < 4; ++j) acc[i][j] = (v8f)(0.0f);

  // thread-constant staging coordinates
  const int r4 = tid >> 3, q4 = tid & 7;   // f32 path: 4-elem chunks, 8 per row, row step 16
  const int r8 = tid >> 2, q8 = tid & 3;   // bf16 path: 8-elem (16B) chunks, 4 per row, row step 32
  const int kk = tid >> 4, cc = tid & 15;  // MODE2 B transpose: 16 chunks of 8 per k-row, k step 8

  auto stageA = [&](int kt, int buf) {
    unsigned short* aB = aLds + buf * (BM * LST);
    if constexpr (MODE == 0 || MODE == 2) {            // f32 -> bf16 while staging
#pragma unroll
      for (int it = 0; it < 8; ++it) {
        const int r = r4 + it * 16;
        const float4 v4 = *(const float4*)(Af + (size_t)(m0 + r) * lda + kt + q4 * 4);
        unsigned short* dst = aB + (size_t)r * LST + q4 * 4;
        *(unsigned int*)(dst + 0) = pack_bf16x2(v4.x, v4.y);
        *(unsigned int*)(dst + 2) = pack_bf16x2(v4.z, v4.w);
      }
    } else {                                           // bf16 source: async 16B copies
#pragma unroll
      for (int it = 0; it < 4; ++it) {
        const int r = r8 + it * 32;
        cp16_g2l(Ah + (size_t)(m0 + r) * lda + kt + q8 * 8, aB + (size_t)r * LST + q8 * 8);
      }
    }
  };
  auto stageB = [&](int kt, int buf) {
    unsigned short* bB = bLds + buf * (BN * LST);
    if constexpr (MODE == 0 || MODE == 3) {            // f32 weight [n][k] -> bf16
#pragma unroll
      for (int it = 0; it < 8; ++it) {
        const int r = r4 + it * 16;
        const float4 v4 = *(const float4*)(Bf + (size_t)(n0 + r) * ldb + kt + q4 * 4);
        unsigned short* dst = bB + (size_t)r * LST + q4 * 4;
        *(unsigned int*)(dst + 0) = pack_bf16x2(v4.x, v4.y);
        *(unsigned int*)(dst + 2) = pack_bf16x2(v4.z, v4.w);
      }
    } else if constexpr (MODE == 1) {                  // bf16 K-matrix [n][k]: async 16B copies
#pragma unroll
      for (int it = 0; it < 4; ++it) {
        const int r = r8 + it * 32;
        cp16_g2l(Bh + (size_t)(n0 + r) * ldb + kt + q8 * 8, bB + (size_t)r * LST + q8 * 8);
      }
    } else {                                           // MODE 2: V[k][n] -> transpose into [n][k]
#pragma unroll
      for (int it = 0; it < 4; ++it) {
        const int k = kk + it * 8;
        const uint4 v4 = *(const uint4*)(Bh + (size_t)(kt + k) * ldb + n0 + cc * 8);
        unsigned short e[8];
        e[0] = (unsigned short)v4.x; e[1] = (unsigned short)(v4.x >> 16);
        e[2] = (unsigned short)v4.y; e[3] = (unsigned short)(v4.y >> 16);
        e[4] = (unsigned short)v4.z; e[5] = (unsigned short)(v4.z >> 16);
        e[6] = (unsigned short)v4.w; e[7] = (unsigned short)(v4.w >> 16);
#pragma unroll
        for (int j = 0; j < 8; ++j) bB[(size_t)(cc * 8 + j) * LST + k] = e[j];
      }
    }
  };
  auto compute = [&](int buf) {
    const unsigned short* aB = aLds + buf * (BM * LST);
    const unsigned short* bB = bLds + buf * (BN * LST);
    v16bf bF[4];
#pragma unroll
    for (int ni = 0; ni < 4; ++ni) bF[ni] = load_frag_b(bB, wn + 16 * ni, lane);
#pragma unroll
    for (int mi = 0; mi < 4; ++mi) {
      const v16bf aF = load_frag_a(aB, wm + 16 * mi, lane);
#pragma unroll
      for (int ni = 0; ni < 4; ++ni)
        acc[mi][ni] = __builtin_amdgcn_wmma_f32_16x16x32_bf16(
            false, aF, false, bF[ni], (short)0, acc[mi][ni], false, false);
    }
  };

  if constexpr (MODE == 1) {
    // fully-async, double-buffered pipeline (8 async ops per wave per tile)
    const int nt = kEnd / BK;
    stageA(0, 0); stageB(0, 0);
    for (int t = 0; t < nt; ++t) {
      const int cur = t & 1;
      if (t + 1 < nt) {
        stageA((t + 1) * BK, cur ^ 1); stageB((t + 1) * BK, cur ^ 1);
        wait_async_prev();             // tile t done; tile t+1 may remain in flight
      } else {
        wait_async_all();
      }
      __syncthreads();
      compute(cur);
      __syncthreads();
    }
  } else {
    for (int kt = 0; kt < kEnd; kt += BK) {
      stageA(kt, 0); stageB(kt, 0);
      if constexpr (MODE == 3) wait_async_all();       // A staged via async path
      __syncthreads();
      compute(0);
      __syncthreads();
    }
  }

  // ---- store (C/D layout: VGPR v -> row v + (lane>>4)*8, col lane&15) ----
#pragma unroll
  for (int mi = 0; mi < 4; ++mi)
#pragma unroll
    for (int ni = 0; ni < 4; ++ni) {
      v8f c = acc[mi][ni];
#pragma unroll
      for (int v = 0; v < 8; ++v) {
        const int row = m0 + wm + mi * 16 + v + ((lane >> 4) << 3);
        const int col = n0 + wn + ni * 16 + (lane & 15);
        float val = c[v] * alpha;
        if constexpr (MODE == 0 || MODE == 3) val += bias[col];
        if constexpr (MODE == 0) {
          const int b = row >> 10, s = row & 1023, h = col >> 7, dk = col & 127;
          ((unsigned short*)Cdst)[(((size_t)(b * H_ + h) * S_) + s) * DK_ + dk] = f32_to_bf16_rne(val);
        } else if constexpr (MODE == 1) {
          ((float*)Cdst)[(size_t)bh * S_ * S_ + (size_t)row * S_ + col] = val;
        } else if constexpr (MODE == 2) {
          const int b = bh >> 3, h = bh & 7;
          ((unsigned short*)Cdst)[((size_t)(b * S_ + row)) * D_ + h * DK_ + col] = f32_to_bf16_rne(val);
        } else {
          ((float*)Cdst)[(size_t)row * D_ + col] = val;
        }
      }
    }
}

// -------------------- row-wise softmax / cumsum / decay pass --------------------
__device__ __forceinline__ float waveReduceMax(float v) {
#pragma unroll
  for (int m = 16; m > 0; m >>= 1) v = fmaxf(v, __shfl_xor(v, m, 32));
  return v;
}
__device__ __forceinline__ float waveReduceSum(float v) {
#pragma unroll
  for (int m = 16; m > 0; m >>= 1) v += __shfl_xor(v, m, 32);
  return v;
}
__device__ __forceinline__ float waveExclScan(float v, int lane) {
  float x = v;
#pragma unroll
  for (int off = 1; off < 32; off <<= 1) {
    float n = __shfl_up(x, off, 32);
    if (lane >= off) x += n;
  }
  return x - v;
}

__global__ void rowpass_kernel(float* __restrict__ attn, const float* __restrict__ gammas) {
  const int wave = threadIdx.x >> 5, lane = threadIdx.x & 31;
  const int rowIdx = blockIdx.x * 8 + wave;     // 0 .. B*H*S-1
  const int bh = rowIdx >> 10;
  const int s  = rowIdx & 1023;
  const int h  = bh & (H_ - 1);
  const float gamma = -fabsf(gammas[h]);
  float* rowp = attn + (size_t)bh * S_ * S_ + (size_t)s * S_;
  const int tbase = lane * 32;

  float raw[32], cum[32];
#pragma unroll
  for (int q = 0; q < 8; ++q) {
    float4 v4 = *(const float4*)(rowp + tbase + q * 4);
    raw[q * 4 + 0] = v4.x; raw[q * 4 + 1] = v4.y; raw[q * 4 + 2] = v4.z; raw[q * 4 + 3] = v4.w;
  }
  float mx = -3.4e38f;
#pragma unroll
  for (int j = 0; j < 32; ++j) { float x = (tbase + j > s) ? NEGV : raw[j]; mx = fmaxf(mx, x); }
  mx = waveReduceMax(mx);
  float sum = 0.f;
#pragma unroll
  for (int j = 0; j < 32; ++j) { float x = (tbase + j > s) ? NEGV : raw[j]; float p = __expf(x - mx); cum[j] = p; sum += p; }
  sum = waveReduceSum(sum);
  const float inv = 1.0f / sum;
  float run = 0.f;
#pragma unroll
  for (int j = 0; j < 32; ++j) { run += cum[j] * inv; cum[j] = run; }
  const float off = waveExclScan(run, lane);
  float mx2 = -3.4e38f;
#pragma unroll
  for (int j = 0; j < 32; ++j) {
    const float distcum = cum[j] + off;         // disttotal == 1 after normalization
    const float rem = 1.0f - distcum;
    const float pe  = fabsf((float)(s - (tbase + j)));
    const float dsc = sqrtf(fmaxf(rem * pe, 0.0f));
    float eff = __expf(dsc * gamma);
    eff = fminf(fmaxf(eff, 1e-5f), 1e5f);
    const float s2 = raw[j] * eff;
    raw[j] = (tbase + j > s) ? NEGV : s2;
    mx2 = fmaxf(mx2, raw[j]);
  }
  mx2 = waveReduceMax(mx2);
  float sum2 = 0.f;
#pragma unroll
  for (int j = 0; j < 32; ++j) { float p = __expf(raw[j] - mx2); raw[j] = p; sum2 += p; }
  sum2 = waveReduceSum(sum2);
  const float inv2 = 1.0f / sum2;
#pragma unroll
  for (int j = 0; j < 32; ++j) raw[j] = (tbase + j > s) ? 0.0f : raw[j] * inv2;
#pragma unroll
  for (int q = 0; q < 8; ++q) {
    float4 v4; v4.x = raw[q * 4 + 0]; v4.y = raw[q * 4 + 1]; v4.z = raw[q * 4 + 2]; v4.w = raw[q * 4 + 3];
    *(float4*)(rowp + tbase + q * 4) = v4;
  }
}

extern "C" void kernel_launch(void* const* d_in, const int* in_sizes, int n_in,
                              void* d_out, int out_size, void* d_ws, size_t ws_size,
                              hipStream_t stream) {
  (void)in_sizes; (void)n_in; (void)out_size; (void)ws_size;
  const float* query  = (const float*)d_in[0];
  const float* key_in = (const float*)d_in[1];
  const float* value  = (const float*)d_in[2];
  // d_in[3] = mask (strict upper triangle) recomputed analytically in-kernel
  const float* Wq = (const float*)d_in[4];
  const float* bq = (const float*)d_in[5];
  const float* Wk = (const float*)d_in[6];
  const float* bk = (const float*)d_in[7];
  const float* Wv = (const float*)d_in[8];
  const float* bv = (const float*)d_in[9];
  const float* Wo = (const float*)d_in[10];
  const float* bo = (const float*)d_in[11];
  const float* gammas = (const float*)d_in[12];

  float* out  = (float*)d_out;
  float* attn = out + (size_t)B_ * S_ * D_;      // [B,H,S,S] region of d_out

  const size_t per = (size_t)B_ * S_ * D_;       // 8,388,608 elements
  unsigned short* q16   = (unsigned short*)d_ws; // [B,H,S,DK] bf16
  unsigned short* k16   = q16 + per;
  unsigned short* v16   = q16 + 2 * per;
  unsigned short* ctx16 = q16 + 3 * per;         // [B,S,D] bf16

  const dim3 blk(NTHREADS);
  const dim3 gProj(D_ / BN, (B_ * S_) / BM);     // 8 x 64
  const dim3 gScore(S_ / BN, S_ / BM, B_ * H_);  // 8 x 8 x 64
  const dim3 gAV(DK_ / BN, S_ / BM, B_ * H_);    // 1 x 8 x 64
  const float inv_sqrt_dk = 0.08838834764831845f;

  gemm_wmma_kernel<0><<<gProj, blk, 0, stream>>>(query,  D_, Wq, D_, bq, q16, D_, 1.0f);
  gemm_wmma_kernel<0><<<gProj, blk, 0, stream>>>(key_in, D_, Wk, D_, bk, k16, D_, 1.0f);
  gemm_wmma_kernel<0><<<gProj, blk, 0, stream>>>(value,  D_, Wv, D_, bv, v16, D_, 1.0f);

  gemm_wmma_kernel<1><<<gScore, blk, 0, stream>>>(q16, DK_, k16, DK_, nullptr, attn, DK_, inv_sqrt_dk);

  rowpass_kernel<<<dim3((B_ * H_ * S_) / 8), dim3(256), 0, stream>>>(attn, gammas);

  gemm_wmma_kernel<2><<<gAV, blk, 0, stream>>>(attn, S_, v16, DK_, nullptr, ctx16, S_, 1.0f);

  gemm_wmma_kernel<3><<<gProj, blk, 0, stream>>>(ctx16, D_, Wo, D_, bo, out, D_, 1.0f);
}
